// TimeSeriesEncEmbedding_23063974380037
// MI455X (gfx1250) — matrix operand order repaired and checked by reading, not compiled, not run on hardware
//
#include <hip/hip_runtime.h>
#include <hip/hip_bf16.h>

// CDNA5 (gfx1250, wave32) WMMA implementation of the per-feature MLP embedding.
// D = A(f16,16x32) x B(f16,32x16) + C(f32) via v_wmma_f32_16x16x32_f16.
// Double-buffered h-tile so GELU VALU for feature f+1 co-executes with the
// WMMAs for feature f (XDL WMMA tracked as TRANS -> VALU co-execution).

typedef __attribute__((ext_vector_type(16))) _Float16 v16h;
typedef __attribute__((ext_vector_type(8)))  _Float16 v8h;
typedef __attribute__((ext_vector_type(8)))  float    v8f;

#define F_  16
#define H_  64
#define E_  256
#define BS_ 65536        // B*S = 32*2048
#define RM  64           // output rows per block
#define HSTRIDE 72       // f16 elems per hTile row (144B: 16B aligned, conflict-free)

__device__ __forceinline__ float gelu_exact(float v) {
    // 0.5*v*(1+erf(v/sqrt(2))) : matches jax.nn.gelu(approximate=False)
    return 0.5f * v * (1.0f + erff(v * 0.70710678118654752f));
}

__global__ __launch_bounds__(256)
void w2_to_f16_kernel(const float* __restrict__ W2, _Float16* __restrict__ W2h) {
    int i = blockIdx.x * 256 + threadIdx.x;       // F*E*H = 262144 elements exactly
    W2h[i] = (_Float16)W2[i];
}

__global__ __launch_bounds__(256)
void tse_embed_kernel(const float* __restrict__ x,            // [BS,F] f32
                      const unsigned char* __restrict__ mask, // [BS,F] bool (1B)
                      const float* __restrict__ W1,           // [F,H]
                      const float* __restrict__ b1,           // [F,H]
                      const _Float16* __restrict__ W2h,       // [F,E,H] f16 (K-contiguous)
                      const float* __restrict__ b2,           // [F,E]
                      const float* __restrict__ mv,           // [F,E] masked_values
                      float* __restrict__ out)                // [BS,E] f32
{
    __shared__ _Float16 hTile[2 * RM * HSTRIDE]; // double-buffered, 18 KB
    __shared__ float    dTile[F_][E_];           // mv - b2, 16 KB
    __shared__ float    colsum[E_];              // sum_f b2[f][col], 1 KB
    __shared__ unsigned rowbits[RM];             // 16-bit mask per row
    __shared__ float    xTile[RM * F_];          // x rows for this block, 4 KB
    __shared__ float    w1T[F_ * H_];            // 4 KB
    __shared__ float    b1T[F_ * H_];            // 4 KB

    const int t      = threadIdx.x;
    const int wave   = t >> 5;                 // 0..7 (wave32)
    const int lane   = t & 31;
    const int l16    = lane & 15;
    const int hiHalf = lane >> 4;              // 0: lanes 0-15, 1: lanes 16-31
    const int r0     = blockIdx.x * RM;

    // ---------------- prologue (once per block) ----------------
    #pragma unroll
    for (int i = 0; i < F_; ++i) {
        const int idx = i * 256 + t;           // covers all F_*E_ = 4096
        dTile[idx >> 8][idx & 255] = mv[idx] - b2[idx];
    }
    {
        float s = 0.0f;
        #pragma unroll
        for (int f = 0; f < F_; ++f) s += b2[f * E_ + t];
        colsum[t] = s;
    }
    // x rows (1024 floats), W1/b1 (1024 floats each) via 16B vector loads
    ((float4*)xTile)[t] = ((const float4*)(x + (size_t)r0 * F_))[t];
    ((float4*)w1T)[t]   = ((const float4*)W1)[t];
    ((float4*)b1T)[t]   = ((const float4*)b1)[t];
    if (t < RM) {
        const unsigned char* mp = mask + (size_t)(r0 + t) * F_;
        unsigned bits = 0;
        #pragma unroll
        for (int f = 0; f < F_; ++f) bits |= (mp[f] ? 1u : 0u) << f;
        rowbits[t] = bits;
    }
    __syncthreads();

    v8f acc[4][2];
    {
        v8f z = {};
        #pragma unroll
        for (int mt = 0; mt < 4; ++mt) { acc[mt][0] = z; acc[mt][1] = z; }
    }

    const int hRow = t >> 2;            // 0..63 : row this thread fills in hTile
    const int hSeg = (t & 3) * 16;      // 0/16/32/48 : H segment

    // GELU producer for feature fI into buffer dst (masked rows -> 0)
    auto computeH = [&](int fI, _Float16* dst) {
        const unsigned masked = (rowbits[hRow] >> fI) & 1u;
        const float xv = xTile[hRow * F_ + fI];
        const float* w1p = w1T + fI * H_ + hSeg;
        const float* b1p = b1T + fI * H_ + hSeg;
        _Float16* hp = dst + hRow * HSTRIDE + hSeg;
        v8h ha, hb;
        #pragma unroll
        for (int j = 0; j < 8; ++j)
            ha[j] = (_Float16)(masked ? 0.0f : gelu_exact(xv * w1p[j] + b1p[j]));
        #pragma unroll
        for (int j = 0; j < 8; ++j)
            hb[j] = (_Float16)(masked ? 0.0f : gelu_exact(xv * w1p[8 + j] + b1p[8 + j]));
        *(v8h*)hp       = ha;
        *(v8h*)(hp + 8) = hb;
    };

    computeH(0, hTile);      // fill buffer 0 with feature 0
    __syncthreads();

    const int kbA = hiHalf * 8;      // A layout: lanes>=16 start at K+8
    const int kbB = hiHalf * 8;      // B layout: lanes>=16 start at K+8 (same as A)

    // ---------------- main loop over features ----------------
    for (int f = 0; f < F_; ++f) {
        const _Float16* cur = hTile + (f & 1) * (RM * HSTRIDE);
        _Float16*       nxt = hTile + ((f + 1) & 1) * (RM * HSTRIDE);

        #pragma unroll
        for (int kk = 0; kk < 2; ++kk) {            // K = kk*32 .. kk*32+31
            // A fragments: 4 M-tiles, documented 16-bit 16x32 lane layout
            v16h a[4];
            #pragma unroll
            for (int mt = 0; mt < 4; ++mt) {
                const _Float16* hr = cur + (mt * 16 + l16) * HSTRIDE + kk * 32;
                v8h lo = *(const v8h*)(hr + kbA);         // K = kk*32 + kbA .. +7
                v8h hi = *(const v8h*)(hr + 16 + kbA);    // K = kk*32 + 16 + kbA .. +7
                a[mt] = __builtin_shufflevector(lo, hi,
                        0,1,2,3,4,5,6,7,8,9,10,11,12,13,14,15);
            }
            // B fragments: 2 N-tiles from global f16 W2 (L2-resident, K-contiguous)
            v16h bf[2];
            #pragma unroll
            for (int nt = 0; nt < 2; ++nt) {
                const int e = wave * 32 + nt * 16 + l16;
                const _Float16* q = W2h + ((size_t)f * E_ + e) * H_ + kk * 32 + kbB;
                v8h lo = *(const v8h*)q;
                v8h hi = *(const v8h*)(q + 16);
                bf[nt] = __builtin_shufflevector(lo, hi,
                        0,1,2,3,4,5,6,7,8,9,10,11,12,13,14,15);
            }
            #pragma unroll
            for (int mt = 0; mt < 4; ++mt)
                #pragma unroll
                for (int nt = 0; nt < 2; ++nt)
                    acc[mt][nt] = __builtin_amdgcn_wmma_f32_16x16x32_f16(
                        false, a[mt], false, bf[nt], (short)0, acc[mt][nt], false, false);
        }

        // Produce h for the next feature in the *other* buffer, in the same
        // barrier interval as the WMMAs above (VALU / XDL co-execution).
        if (f + 1 < F_) {
            computeH(f + 1, nxt);
            __syncthreads();
        }
    }

    // ---------------- epilogue: + sum_f b2  + sum_{masked f} (mv-b2), store ----------------
    const int col0 = wave * 32 + l16;
    const int col1 = col0 + 16;
    float dreg0[F_], dreg1[F_];
    #pragma unroll
    for (int f = 0; f < F_; ++f) { dreg0[f] = dTile[f][col0]; dreg1[f] = dTile[f][col1]; }
    const float cs0 = colsum[col0], cs1 = colsum[col1];

    #pragma unroll
    for (int mt = 0; mt < 4; ++mt) {
        #pragma unroll
        for (int r = 0; r < 8; ++r) {
            const int rowT = mt * 16 + r + hiHalf * 8;    // C layout: VGPR r -> M=r (+8 hi lanes)
            const unsigned bits = rowbits[rowT];
            float add0 = cs0, add1 = cs1;
            #pragma unroll
            for (int f = 0; f < F_; ++f) {
                const float m = (float)((bits >> f) & 1u);
                add0 += m * dreg0[f];
                add1 += m * dreg1[f];
            }
            const size_t go = (size_t)(r0 + rowT) * E_;
            out[go + col0] = acc[mt][0][r] + add0;
            out[go + col1] = acc[mt][1][r] + add1;
        }
    }
}

extern "C" void kernel_launch(void* const* d_in, const int* in_sizes, int n_in,
                              void* d_out, int out_size, void* d_ws, size_t ws_size,
                              hipStream_t stream) {
    (void)in_sizes; (void)n_in; (void)out_size; (void)ws_size;
    const float*         x    = (const float*)d_in[0];
    const unsigned char* msk  = (const unsigned char*)d_in[1];  // jax bool -> 1 byte
    const float*         W1   = (const float*)d_in[2];
    const float*         b1   = (const float*)d_in[3];
    const float*         W2   = (const float*)d_in[4];
    const float*         b2   = (const float*)d_in[5];
    const float*         mv   = (const float*)d_in[6];
    float*               out  = (float*)d_out;

    _Float16* W2h = (_Float16*)d_ws;   // needs F*E*H*2 = 512 KB of workspace

    w2_to_f16_kernel<<<(F_ * E_ * H_) / 256, 256, 0, stream>>>(W2, W2h);
    tse_embed_kernel<<<BS_ / RM, 256, 0, stream>>>(x, msk, W1, b1, W2h, b2, mv, out);
}
